// DotProductAttention_48395691491699
// MI455X (gfx1250) — matrix operand order repaired and checked by reading, not compiled, not run on hardware
//
#include <hip/hip_runtime.h>

// ---------------------------------------------------------------------------
// Types for CDNA5 WMMA
// ---------------------------------------------------------------------------
typedef __attribute__((ext_vector_type(16))) __bf16 v16bf;
typedef __attribute__((ext_vector_type(8)))  float  v8f;

// fp32 -> bf16 (round to nearest even)
__device__ __forceinline__ unsigned short f32_to_bf16(float f) {
    unsigned int x = __float_as_uint(f);
    x += 0x7FFFu + ((x >> 16) & 1u);
    return (unsigned short)(x >> 16);
}

// ---------------------------------------------------------------------------
// Fragment loaders (ISA 7.12.2 layouts, wave32), vectorized.
// A 16x32 bf16:  lane L -> row m=L&15, two contiguous K runs:
//                k in [8h, 8h+8) and [16+8h, 24+8h), h=L>>4   -> 2x b128
// B 32x16 bf16:  lane L -> col n=L&15, two contiguous K runs:
//                k in [8h, 8h+8) and [16+8h, 24+8h), h=L>>4   -> 2x b128
// C/D 16x16 f32: vgpr i -> row m = i + 8h, col n = L&15
// ---------------------------------------------------------------------------

// A fragment; element (m,k) at base[m*ld + k]. ld (ushorts) multiple of 8,
// base 16B-aligned  -> two 16B loads per lane.
__device__ __forceinline__ v16bf load_a_frag_vec(const unsigned short* base, int ld) {
    union { v16bf v; uint4 q[2]; } f;
    const int lane = threadIdx.x & 31;
    const int m = lane & 15;
    const int hoff = (lane >> 4) * 8;
    f.q[0] = *(const uint4*)(base + m * ld + hoff);
    f.q[1] = *(const uint4*)(base + m * ld + 16 + hoff);
    return f.v;
}

// A fragment from fp32 source (converted); element (m,k) at base[m*ld + k].
__device__ __forceinline__ v16bf load_a_frag_f32_vec(const float* base, int ld) {
    union { v16bf v; unsigned short u[16]; } f;
    const int lane = threadIdx.x & 31;
    const int m = lane & 15;
    const int hoff = (lane >> 4) * 8;
    const float* p0 = base + m * ld + hoff;
    const float* p1 = base + m * ld + 16 + hoff;
    float4 a0 = *(const float4*)(p0);
    float4 a1 = *(const float4*)(p0 + 4);
    float4 b0 = *(const float4*)(p1);
    float4 b1 = *(const float4*)(p1 + 4);
    f.u[0] = f32_to_bf16(a0.x); f.u[1] = f32_to_bf16(a0.y);
    f.u[2] = f32_to_bf16(a0.z); f.u[3] = f32_to_bf16(a0.w);
    f.u[4] = f32_to_bf16(a1.x); f.u[5] = f32_to_bf16(a1.y);
    f.u[6] = f32_to_bf16(a1.z); f.u[7] = f32_to_bf16(a1.w);
    f.u[8]  = f32_to_bf16(b0.x); f.u[9]  = f32_to_bf16(b0.y);
    f.u[10] = f32_to_bf16(b0.z); f.u[11] = f32_to_bf16(b0.w);
    f.u[12] = f32_to_bf16(b1.x); f.u[13] = f32_to_bf16(b1.y);
    f.u[14] = f32_to_bf16(b1.z); f.u[15] = f32_to_bf16(b1.w);
    return f.v;
}

// B fragment from K-contiguous storage: element (k,n) at base[n*ld + k].
// ld (ushorts) multiple of 8, base 16B-aligned -> two 16B loads per lane.
__device__ __forceinline__ v16bf load_b_frag_kcontig(const unsigned short* base, int ld) {
    union { v16bf v; uint4 q[2]; } f;
    const int lane = threadIdx.x & 31;
    const int n = lane & 15;
    const int koff = (lane >> 4) * 8;
    f.q[0] = *(const uint4*)(base + n * ld + koff);
    f.q[1] = *(const uint4*)(base + n * ld + 16 + koff);
    return f.v;
}

#define WMMA_BF16(a, b, c) \
    __builtin_amdgcn_wmma_f32_16x16x32_bf16(false, (a), false, (b), (short)0, (c), false, false)

// ---------------------------------------------------------------------------
// gfx1250 async global->LDS copy (ASYNCcnt-tracked), ISA 15.18.3 op 98.
// lds_off: raw LDS byte address (low 32 bits of generic shared pointer).
// ---------------------------------------------------------------------------
__device__ __forceinline__ void async_load_b128(unsigned lds_off, const void* gaddr) {
    asm volatile("global_load_async_to_lds_b128 %0, %1, off"
                 :: "v"(lds_off), "v"(gaddr) : "memory");
}
__device__ __forceinline__ void wait_async0() {
    asm volatile("s_wait_asynccnt 0" ::: "memory");
}

// ---------------------------------------------------------------------------
// Kernel 1: fused QKV projection.  C[4096,1024] = X[4096,1024]*W[1024,1024]
// + bias; Q scaled by H^-0.5.  bf16 out.  grid(16,64,3), block 128 (4 waves).
// W tile stored TRANSPOSED in LDS so B-fragment loads are K-contiguous b128s.
// ---------------------------------------------------------------------------
__global__ __launch_bounds__(128) void qkv_proj_kernel(
    const float* __restrict__ xq, const float* __restrict__ xk, const float* __restrict__ xv,
    const float* __restrict__ wq, const float* __restrict__ wk, const float* __restrict__ wv,
    const float* __restrict__ bq, const float* __restrict__ bk, const float* __restrict__ bv,
    unsigned short* __restrict__ oq, unsigned short* __restrict__ ok, unsigned short* __restrict__ ov)
{
    __shared__ unsigned short As[64][40];   // A tile 64x32, row-major, ld 40
    __shared__ unsigned short Bst[64][40];  // W tile 32x64 stored [n][k], ld 40

    const int z = blockIdx.z;
    const float* X  = (z == 0) ? xq : (z == 1) ? xk : xv;
    const float* W  = (z == 0) ? wq : (z == 1) ? wk : wv;
    const float* Bi = (z == 0) ? bq : (z == 1) ? bk : bv;
    unsigned short* O = (z == 0) ? oq : (z == 1) ? ok : ov;
    const float scale = (z == 0) ? 0.125f : 1.0f;

    const int n0 = blockIdx.x * 64;
    const int m0 = blockIdx.y * 64;
    const int tid = threadIdx.x;
    const int wid = tid >> 5;
    const int lane = tid & 31;

    v8f zero = {0.f, 0.f, 0.f, 0.f, 0.f, 0.f, 0.f, 0.f};
    v8f acc[4] = {zero, zero, zero, zero};

    for (int k0 = 0; k0 < 1024; k0 += 32) {
        __syncthreads();
#pragma unroll
        for (int e = 0; e < 16; ++e) {
            const int idx = tid + e * 128;
            const int r  = idx >> 5, c  = idx & 31;     // A: 64x32
            As[r][c] = f32_to_bf16(X[(size_t)(m0 + r) * 1024 + k0 + c]);
            const int r2 = idx >> 6, c2 = idx & 63;     // W: 32x64, transpose into LDS
            Bst[c2][r2] = f32_to_bf16(W[(size_t)(k0 + r2) * 1024 + n0 + c2]);
        }
        __syncthreads();

        const v16bf a = load_a_frag_vec(&As[wid * 16][0], 40);
#pragma unroll
        for (int t = 0; t < 4; ++t) {
            const v16bf b = load_b_frag_kcontig(&Bst[t * 16][0], 40);
            acc[t] = WMMA_BF16(a, b, acc[t]);
        }
    }

    const int hf = lane >> 4;
#pragma unroll
    for (int t = 0; t < 4; ++t) {
#pragma unroll
        for (int i = 0; i < 8; ++i) {
            const int row = m0 + wid * 16 + i + 8 * hf;
            const int col = n0 + t * 16 + (lane & 15);
            const float v = (acc[t][i] + Bi[col]) * scale;
            O[(size_t)row * 1024 + col] = f32_to_bf16(v);
        }
    }
}

// ---------------------------------------------------------------------------
// Kernel 2: attention per (b, head, 16-row q tile).  grid(128,16,2), block 256.
// K tile staged via async global->LDS b128 (row-major [s][h] is already
// K-contiguous for the QK^T B fragment).  V tile staged transposed [h][s].
// dynamic LDS = 131072 (logits) + 2048 (q) + 17408 (k/v) = 150528 bytes.
// ---------------------------------------------------------------------------
__global__ __launch_bounds__(256) void attn_kernel(
    const unsigned short* __restrict__ qb,
    const unsigned short* __restrict__ kb,
    const unsigned short* __restrict__ vb,
    unsigned short* __restrict__ enc,
    float* __restrict__ probs)
{
    extern __shared__ char smem[];
    float*          Lg = (float*)smem;                             // [16][2048] fp32
    unsigned short* qs = (unsigned short*)(smem + 131072);         // [16][64]
    unsigned short* kv = (unsigned short*)(smem + 131072 + 2048);  // K:[128][64] / V:[64][136]

    const int m0 = blockIdx.x * 16;
    const int n  = blockIdx.y;
    const int b  = blockIdx.z;
    const int tid = threadIdx.x;
    const int wid = tid >> 5;
    const int lane = tid & 31;
    const int hf = lane >> 4;
    const size_t rowbase = (size_t)b * 2048;
    const unsigned kv_lds = (unsigned)(size_t)(void*)kv;   // raw LDS byte address

    // stage Q tile (16x64 bf16)
#pragma unroll
    for (int e = 0; e < 4; ++e) {
        const int idx = tid + e * 256;
        const int r = idx >> 6, c = idx & 63;
        qs[idx] = qb[(rowbase + m0 + r) * 1024 + n * 64 + c];
    }
    __syncthreads();
    const v16bf aq0 = load_a_frag_vec(qs, 64);       // H 0..31
    const v16bf aq1 = load_a_frag_vec(qs + 32, 64);  // H 32..63

    v8f zero = {0.f, 0.f, 0.f, 0.f, 0.f, 0.f, 0.f, 0.f};

    // ---- logits = softcap(Q K^T); each wave owns one 16-col s-subtile ----
    for (int s0 = 0; s0 < 2048; s0 += 128) {
        __syncthreads();
        // async-stage K tile 128x64 bf16 (16 KB) as 1024 x 16B chunks
#pragma unroll
        for (int e = 0; e < 4; ++e) {
            const int idx = tid + e * 256;       // chunk id
            const int r = idx >> 3, c8 = idx & 7; // row s, 8-ushort chunk in row
            async_load_b128(kv_lds + (unsigned)idx * 16u,
                            kb + (rowbase + s0 + r) * 1024 + n * 64 + c8 * 8);
        }
        wait_async0();
        __syncthreads();

        // B(k=h, n=s_local) = kv[s_local*64 + h] : K-contiguous per lane
        const v16bf b0 = load_b_frag_kcontig(&kv[(wid * 16) * 64], 64);
        const v16bf b1 = load_b_frag_kcontig(&kv[(wid * 16) * 64 + 32], 64);
        v8f c = zero;
        c = WMMA_BF16(aq0, b0, c);
        c = WMMA_BF16(aq1, b1, c);

#pragma unroll
        for (int i = 0; i < 8; ++i) {
            const int m   = i + 8 * hf;
            const int col = s0 + wid * 16 + (lane & 15);
            Lg[m * 2048 + col] = 50.0f * tanhf(c[i] * 0.02f);
        }
    }
    __syncthreads();

    // ---- softmax over S (16 lanes per row; half-wave shuffle reductions) ----
    {
        const int row = tid >> 4;
        const int l16 = tid & 15;
        float mx = -3.4e38f;
        for (int j = l16; j < 2048; j += 16) mx = fmaxf(mx, Lg[row * 2048 + j]);
#pragma unroll
        for (int o = 8; o > 0; o >>= 1) mx = fmaxf(mx, __shfl_xor(mx, o));
        float sum = 0.f;
        for (int j = l16; j < 2048; j += 16) {
            const float e = __expf(Lg[row * 2048 + j] - mx);
            Lg[row * 2048 + j] = e;
            sum += e;
        }
#pragma unroll
        for (int o = 8; o > 0; o >>= 1) sum += __shfl_xor(sum, o);
        const float inv = 1.0f / sum;
        const size_t pbase = (((size_t)b * 16 + n) * 2048 + (m0 + row)) * 2048;
        for (int j = l16; j < 2048; j += 16) {
            const float p = Lg[row * 2048 + j] * inv;
            Lg[row * 2048 + j] = p;       // keep normalized probs for P*V
            probs[pbase + j] = p;         // second reference output
        }
    }
    __syncthreads();

    // ---- encoded = P * V ; waves 0..3 each own a 16-wide slice of H ----
    v8f acc = zero;
    for (int s0 = 0; s0 < 2048; s0 += 128) {
        __syncthreads();
        // stage V tile 128x64 transposed -> vt[h][s], ld 136 (pad, 16B-aligned rows)
#pragma unroll
        for (int e = 0; e < 32; ++e) {
            const int idx = tid + e * 256;
            const int r = idx >> 6, c = idx & 63;   // r = s_local, c = h
            kv[c * 136 + r] = vb[(rowbase + s0 + r) * 1024 + n * 64 + c];
        }
        __syncthreads();
        if (wid < 4) {
#pragma unroll
            for (int qq = 0; qq < 4; ++qq) {
                const v16bf a   = load_a_frag_f32_vec(Lg + (s0 + qq * 32), 2048);
                const v16bf bvf = load_b_frag_kcontig(&kv[(wid * 16) * 136 + qq * 32], 136);
                acc = WMMA_BF16(a, bvf, acc);
            }
        }
    }
    if (wid < 4) {
#pragma unroll
        for (int i = 0; i < 8; ++i) {
            const int m = i + 8 * hf;
            const int h = wid * 16 + (lane & 15);
            enc[(rowbase + m0 + m) * 1024 + n * 64 + h] = f32_to_bf16(acc[i]);
        }
    }
}

// ---------------------------------------------------------------------------
// Kernel 3: output projection. out[4096,1024] = enc_bf16 * wo + bo.
// grid (16, 64), block 128.  W tile transposed in LDS (as in kernel 1).
// ---------------------------------------------------------------------------
__global__ __launch_bounds__(128) void out_proj_kernel(
    const unsigned short* __restrict__ enc,
    const float* __restrict__ wo,
    const float* __restrict__ bo,
    float* __restrict__ out)
{
    __shared__ unsigned short As[64][40];
    __shared__ unsigned short Bst[64][40];

    const int n0 = blockIdx.x * 64;
    const int m0 = blockIdx.y * 64;
    const int tid = threadIdx.x;
    const int wid = tid >> 5;
    const int lane = tid & 31;

    v8f zero = {0.f, 0.f, 0.f, 0.f, 0.f, 0.f, 0.f, 0.f};
    v8f acc[4] = {zero, zero, zero, zero};

    for (int k0 = 0; k0 < 1024; k0 += 32) {
        __syncthreads();
#pragma unroll
        for (int e = 0; e < 16; ++e) {
            const int idx = tid + e * 128;
            const int r  = idx >> 5, c  = idx & 31;
            As[r][c] = enc[(size_t)(m0 + r) * 1024 + k0 + c];
            const int r2 = idx >> 6, c2 = idx & 63;
            Bst[c2][r2] = f32_to_bf16(wo[(size_t)(k0 + r2) * 1024 + n0 + c2]);
        }
        __syncthreads();

        const v16bf a = load_a_frag_vec(&As[wid * 16][0], 40);
#pragma unroll
        for (int t = 0; t < 4; ++t) {
            const v16bf b = load_b_frag_kcontig(&Bst[t * 16][0], 40);
            acc[t] = WMMA_BF16(a, b, acc[t]);
        }
    }

    const int hf = lane >> 4;
#pragma unroll
    for (int t = 0; t < 4; ++t) {
#pragma unroll
        for (int i = 0; i < 8; ++i) {
            const int row = m0 + wid * 16 + i + 8 * hf;
            const int col = n0 + t * 16 + (lane & 15);
            out[(size_t)row * 1024 + col] = acc[t][i] + bo[col];
        }
    }
}

// ---------------------------------------------------------------------------
// Launch
// ---------------------------------------------------------------------------
extern "C" void kernel_launch(void* const* d_in, const int* in_sizes, int n_in,
                              void* d_out, int out_size, void* d_ws, size_t ws_size,
                              hipStream_t stream) {
    (void)in_sizes; (void)n_in; (void)out_size; (void)ws_size;

    const float* xq = (const float*)d_in[0];
    const float* xk = (const float*)d_in[1];
    const float* xv = (const float*)d_in[2];
    const float* wq = (const float*)d_in[3];
    const float* bq = (const float*)d_in[4];
    const float* wk = (const float*)d_in[5];
    const float* bk = (const float*)d_in[6];
    const float* wv = (const float*)d_in[7];
    const float* bv = (const float*)d_in[8];
    const float* wo = (const float*)d_in[9];
    const float* bo = (const float*)d_in[10];

    float* out   = (float*)d_out;                 // [B,T,D]   = 4194304 floats
    float* probs = out + (size_t)4194304;         // [B,N,T,S] = 134217728 floats

    char* ws = (char*)d_ws;                       // 32 MB used
    unsigned short* qb  = (unsigned short*)(ws);                 // 8 MB
    unsigned short* kb  = (unsigned short*)(ws + (8u << 20));    // 8 MB
    unsigned short* vb  = (unsigned short*)(ws + (16u << 20));   // 8 MB
    unsigned short* enc = (unsigned short*)(ws + (24u << 20));   // 8 MB

    // 1) fused QKV projections
    qkv_proj_kernel<<<dim3(16, 64, 3), 128, 0, stream>>>(
        xq, xk, xv, wq, wk, wv, bq, bk, bv, qb, kb, vb);

    // 2) attention (softcap + softmax + probs out + P*V)
    const size_t smem = 131072 + 2048 + 17408;    // 150528 B (< 320 KB/WG)
    attn_kernel<<<dim3(128, 16, 2), 256, smem, stream>>>(qb, kb, vb, enc, probs);

    // 3) output projection
    out_proj_kernel<<<dim3(16, 64), 128, 0, stream>>>(enc, wo, bo, out);
}
